// BSGMP_57045755625634
// MI455X (gfx1250) — compile-verified, matrix operand order checked by reading, not run on hardware
//
#include <hip/hip_runtime.h>

typedef _Float16 f16;
typedef _Float16 v8h  __attribute__((ext_vector_type(8)));
typedef _Float16 v16h __attribute__((ext_vector_type(16)));
typedef float    v8f  __attribute__((ext_vector_type(8)));

#define LAT   128
#define KPMAX 288
#define MT    32          // rows per block (two 16-row WMMA subtiles)

// ---------------------------------------------------------------- utility kernels

__global__ void k_fill(float* __restrict__ p, float v, int n) {
    int t = blockIdx.x * blockDim.x + threadIdx.x;
    if (t < n) p[t] = v;
}

// W [K,128] f32 row-major  ->  Wt [128, KPMAX] f16 (zero padded)
__global__ void k_wconv(const float* __restrict__ W, f16* __restrict__ dst, int K) {
    int t = blockIdx.x * blockDim.x + threadIdx.x;
    if (t >= LAT * KPMAX) return;
    int n = t / KPMAX, k = t - n * KPMAX;
    float v = (k < K) ? W[(size_t)k * LAT + n] : 0.0f;
    dst[(size_t)n * KPMAX + k] = (f16)v;
}

__global__ void k_deg(const int* __restrict__ gi, float* __restrict__ deg, int E) {
    int e = blockIdx.x * blockDim.x + threadIdx.x;
    if (e < E) atomicAdd(&deg[gi[e]], 1.0f);
}

__global__ void k_ws(const int* __restrict__ gi, const int* __restrict__ gj,
                     const float* __restrict__ w, const float* __restrict__ deg,
                     float* __restrict__ wse, float* __restrict__ waggr, int E) {
    int e = blockIdx.x * blockDim.x + threadIdx.x;
    if (e >= E) return;
    int i = gi[e];
    float v = w[i] / fmaxf(deg[i], 1.0f);
    wse[e] = v;
    atomicAdd(&waggr[gj[e]], v);
}

__global__ void k_ec(const float* __restrict__ wse, const float* __restrict__ waggr,
                     const int* __restrict__ gj, float* __restrict__ ec, int E) {
    int e = blockIdx.x * blockDim.x + threadIdx.x;
    if (e < E) ec[e] = wse[e] / fmaxf(waggr[gj[e]], 1e-12f);
}

// out[dst[e]] += x[src[e]] * ew[e]   (C columns)
__global__ void k_econv(const float* __restrict__ x, const int* __restrict__ si,
                        const int* __restrict__ di, const float* __restrict__ ew,
                        float* __restrict__ out, int E, int C) {
    int t = blockIdx.x * blockDim.x + threadIdx.x;
    if (t >= E * C) return;
    int e = t / C, c = t - e * C;
    atomicAdd(&out[(size_t)di[e] * C + c], x[(size_t)si[e] * C + c] * ew[e]);
}

__global__ void k_gather(const float* __restrict__ src, const int* __restrict__ idx,
                         float* __restrict__ dst, int n, int C) {
    int t = blockIdx.x * blockDim.x + threadIdx.x;
    if (t >= n * C) return;
    int r = t / C, c = t - r * C;
    dst[(size_t)r * C + c] = src[(size_t)idx[r] * C + c];
}

__global__ void k_scatter_rows(const float* __restrict__ src, const int* __restrict__ idx,
                               float* __restrict__ dst, int n, int C) {
    int t = blockIdx.x * blockDim.x + threadIdx.x;
    if (t >= n * C) return;
    int r = t / C, c = t - r * C;
    dst[(size_t)idx[r] * C + c] = src[(size_t)r * C + c];
}

// ---------------------------------------------------------------- fused WMMA MLP layer
// One block = 32 output rows x 128 cols.  8 waves; each wave owns one 16-col tile
// and two 16-row subtiles (B fragment reused for 2 WMMAs per K step).
// mode: 0 = f16 activation src [M,KP]; 1 = edge-feature gather (K=288);
//       2 = node concat [x,aggr] (K=256)
// epi : 0 = bias+ReLU -> f16 out; 1 = bias+LN -> atomic scatter-add to scat_out[gi[row]]
//       2 = bias+LN + resid (+resid2) -> f32 out
__global__ __launch_bounds__(256)
void k_mlp(const f16* __restrict__ actsrc,
           const float* __restrict__ xsrc, const float* __restrict__ aux,
           const int* __restrict__ gi, const int* __restrict__ gj,
           const f16* __restrict__ wt, const float* __restrict__ bias,
           const float* __restrict__ lng, const float* __restrict__ lnb,
           float* __restrict__ scat_out,
           const float* __restrict__ resid, const float* __restrict__ resid2,
           float* __restrict__ outf32, f16* __restrict__ outf16,
           int M, int KP, int mode, int epi)
{
    __shared__ f16   Ap[MT * KPMAX];                 // 18432 B
    __shared__ float Cp[MT * LAT];                   // 16384 B
    __shared__ float s_sum[MT * 8], s_sq[MT * 8], s_mu[MT], s_rs[MT];

    const int tid   = threadIdx.x;
    const int tileM = blockIdx.x;
    const int row   = tid >> 3;          // 0..31
    const int s     = tid & 7;           // 8 threads per row
    const int gr    = tileM * MT + row;
    const bool rv   = (gr < M);

    // ---- stage A panel (32 x KP, f16) into LDS
    if (mode == 0) {
        for (int c = s; c < KP; c += 8)
            Ap[row * KP + c] = rv ? actsrc[(size_t)gr * KP + c] : (f16)0.0f;
    } else if (mode == 1) {
        int ii = 0, jj = 0;
        float d0 = 0.f, d1 = 0.f, d2 = 0.f, nr = 0.f;
        if (rv) {
            ii = gi[gr]; jj = gj[gr];
            d0 = aux[ii * 3 + 0] - aux[jj * 3 + 0];
            d1 = aux[ii * 3 + 1] - aux[jj * 3 + 1];
            d2 = aux[ii * 3 + 2] - aux[jj * 3 + 2];
            nr = sqrtf(d0 * d0 + d1 * d1 + d2 * d2);
        }
        for (int c = s; c < KP; c += 8) {
            float v = 0.0f;
            if (rv) {
                if      (c < 128) v = xsrc[(size_t)ii * LAT + c];
                else if (c < 256) v = xsrc[(size_t)jj * LAT + (c - 128)];
                else if (c == 256) v = d0;
                else if (c == 257) v = d1;
                else if (c == 258) v = d2;
                else if (c == 259) v = nr;
            }
            Ap[row * KP + c] = (f16)v;
        }
    } else {
        for (int c = s; c < KP; c += 8) {
            float v = 0.0f;
            if (rv) v = (c < 128) ? xsrc[(size_t)gr * LAT + c]
                                  : aux[(size_t)gr * LAT + (c - 128)];
            Ap[row * KP + c] = (f16)v;
        }
    }
    __syncthreads();

    // ---- WMMA GEMM: each wave owns one 16-col tile, two row subtiles
    const int lane = tid & 31;
    const int wv   = tid >> 5;           // 0..7 column block
    const int n16  = lane & 15;
    const int hi   = lane >> 4;
    const int col  = wv * 16 + n16;

    v8f acc0 = {}, acc1 = {};
    const f16* wrow = wt + (size_t)col * KPMAX;
    for (int k0 = 0; k0 < KP; k0 += 32) {
        // B fragment: lane col n, halves 0..7 -> K=k0+8*hi.., 8..15 -> K=k0+16+8*hi..
        v8h bl = *(const v8h*)&wrow[k0 + 8 * hi];
        v8h bh = *(const v8h*)&wrow[k0 + 16 + 8 * hi];
        // A fragments for row subtiles 0 and 1 (same K pattern, lane row m = n16)
        v8h a0l = *(const v8h*)&Ap[n16 * KP + k0 + 8 * hi];
        v8h a0h = *(const v8h*)&Ap[n16 * KP + k0 + 16 + 8 * hi];
        v8h a1l = *(const v8h*)&Ap[(16 + n16) * KP + k0 + 8 * hi];
        v8h a1h = *(const v8h*)&Ap[(16 + n16) * KP + k0 + 16 + 8 * hi];
        v16h a0, a1, b;
#pragma unroll
        for (int q = 0; q < 8; ++q) {
            b[q]  = bl[q];  b[q + 8]  = bh[q];
            a0[q] = a0l[q]; a0[q + 8] = a0h[q];
            a1[q] = a1l[q]; a1[q + 8] = a1h[q];
        }
        acc0 = __builtin_amdgcn_wmma_f32_16x16x32_f16(false, a0, false, b,
                                                      (short)0, acc0, false, false);
        acc1 = __builtin_amdgcn_wmma_f32_16x16x32_f16(false, a1, false, b,
                                                      (short)0, acc1, false, false);
    }

    const float bc = bias[col];

    if (epi == 0) {                         // bias + ReLU -> f16 activations
#pragma unroll
        for (int r = 0; r < 8; ++r) {
            int m  = r + 8 * hi;
            int g0 = tileM * MT + m;
            int g1 = tileM * MT + 16 + m;
            if (g0 < M) {
                float v = acc0[r] + bc;
                outf16[(size_t)g0 * LAT + col] = (f16)(v > 0.0f ? v : 0.0f);
            }
            if (g1 < M) {
                float v = acc1[r] + bc;
                outf16[(size_t)g1 * LAT + col] = (f16)(v > 0.0f ? v : 0.0f);
            }
        }
        return;
    }

    // ---- LayerNorm epilogues: stage C panel to LDS
#pragma unroll
    for (int r = 0; r < 8; ++r) {
        int m = r + 8 * hi;
        Cp[m * LAT + col]        = acc0[r] + bc;
        Cp[(16 + m) * LAT + col] = acc1[r] + bc;
    }
    __syncthreads();
    {
        float sm = 0.f, sq = 0.f;
#pragma unroll
        for (int q = 0; q < 16; ++q) {
            float v = Cp[row * LAT + s * 16 + q];
            sm += v; sq += v * v;
        }
        s_sum[row * 8 + s] = sm;
        s_sq [row * 8 + s] = sq;
    }
    __syncthreads();
    if (s == 0) {
        float sm = 0.f, sq = 0.f;
        for (int q = 0; q < 8; ++q) { sm += s_sum[row * 8 + q]; sq += s_sq[row * 8 + q]; }
        float mu  = sm * (1.0f / 128.0f);
        float var = sq * (1.0f / 128.0f) - mu * mu;
        s_mu[row] = mu;
        s_rs[row] = rsqrtf(var + 1e-5f);
    }
    __syncthreads();
    if (rv) {
        float mu = s_mu[row], rs = s_rs[row];
        if (epi == 1) {                     // edge-MLP tail: LN + segment_sum to receivers
            int dst = gi[gr];
#pragma unroll
            for (int q = 0; q < 16; ++q) {
                int c = s * 16 + q;
                float y = (Cp[row * LAT + c] - mu) * rs * lng[c] + lnb[c];
                atomicAdd(&scat_out[(size_t)dst * LAT + c], y);
            }
        } else {                            // node-MLP tail: LN + residual(s)
#pragma unroll
            for (int q = 0; q < 16; ++q) {
                int c = s * 16 + q;
                float y = (Cp[row * LAT + c] - mu) * rs * lng[c] + lnb[c];
                y += resid[(size_t)gr * LAT + c];
                if (resid2) y += resid2[(size_t)gr * LAT + c];
                outf32[(size_t)gr * LAT + c] = y;
            }
        }
    }
}

// ---------------------------------------------------------------- host orchestration

extern "C" void kernel_launch(void* const* d_in, const int* in_sizes, int n_in,
                              void* d_out, int out_size, void* d_ws, size_t ws_size,
                              hipStream_t stream)
{
    (void)n_in; (void)out_size; (void)ws_size;

    // ---- input layout (jax pytree order: params dict keys sorted)
    const float* h_in   = (const float*)d_in[0];
    const float* pos_in = (const float*)d_in[1];
    // params leaves [2..81]: bottom, down0, down1, up0, up1 (16 leaves each:
    //   e.W0,e.b0,e.W1,e.b1,e.W2,e.b2,e.ln_b,e.ln_g, n.W0..n.ln_g)
    const int BOT = 2, D0 = 18, D1 = 34, U0 = 50, U1 = 66;
    const int exec_base[5] = { D0, D1, BOT, U0, U1 };   // down0,down1,bottom,up0,up1

    const int* gs[3]; const int* mids[2];
    for (int l = 0; l < 3; ++l) gs[l] = (const int*)d_in[82 + l];
    mids[0] = (const int*)d_in[85];
    mids[1] = (const int*)d_in[86];

    const int N0 = in_sizes[0] / LAT;
    int Ecnt[3]  = { in_sizes[82] / 2, in_sizes[83] / 2, in_sizes[84] / 2 };
    int Nlv[3]   = { N0, in_sizes[85], in_sizes[86] };
    int Emax = Ecnt[0];
    for (int l = 1; l < 3; ++l) if (Ecnt[l] > Emax) Emax = Ecnt[l];

    auto P = [&](int idx) { return (const float*)d_in[idx]; };

    // ---- workspace carve
    char* w8 = (char*)d_ws;
    size_t off = 0;
    auto carve = [&](size_t bytes) -> void* {
        void* p = w8 + off;
        off += (bytes + 255) & ~(size_t)255;
        return p;
    };
    f16*   wt    = (f16*)carve((size_t)30 * LAT * KPMAX * sizeof(f16));
    f16*   act0  = (f16*)carve((size_t)Emax * LAT * sizeof(f16));
    f16*   act1  = (f16*)carve((size_t)Emax * LAT * sizeof(f16));
    float* aggr  = (float*)carve((size_t)N0 * LAT * 4);
    float* hA    = (float*)carve((size_t)N0 * LAT * 4);
    float* hB    = (float*)carve((size_t)N0 * LAT * 4);
    float* hU    = (float*)carve((size_t)N0 * LAT * 4);
    float* dout0 = (float*)carve((size_t)Nlv[0] * LAT * 4);
    float* dout1 = (float*)carve((size_t)Nlv[1] * LAT * 4);
    float* posA  = (float*)carve((size_t)N0 * 3 * 4);
    float* posB  = (float*)carve((size_t)N0 * 3 * 4);
    float* dpos0 = (float*)carve((size_t)Nlv[0] * 3 * 4);
    float* dpos1 = (float*)carve((size_t)Nlv[1] * 3 * 4);
    float* wcur  = (float*)carve((size_t)N0 * 4);
    float* waggr = (float*)carve((size_t)N0 * 4);
    float* deg   = (float*)carve((size_t)N0 * 4);
    float* wse   = (float*)carve((size_t)Emax * 4);
    float* ct0   = (float*)carve((size_t)Ecnt[0] * 4);
    float* ct1   = (float*)carve((size_t)Ecnt[1] * 4);
    float* cts[2]   = { ct0, ct1 };
    float* dposv[2] = { dpos0, dpos1 };
    float* doutv[2] = { dout0, dout1 };

    auto gsz = [](int n) { return (n + 255) / 256; };

    // ---- convert all 30 weight matrices to transposed, padded f16
    for (int ge = 0; ge < 5; ++ge) {
        int base = exec_base[ge];
        for (int mlp = 0; mlp < 2; ++mlp)
            for (int ly = 0; ly < 3; ++ly) {
                int widx = base + mlp * 8 + ly * 2;
                int K = (ly == 0) ? (mlp == 0 ? (2 * LAT + 4) : (2 * LAT)) : LAT;
                f16* dst = wt + (size_t)(ge * 6 + mlp * 3 + ly) * LAT * KPMAX;
                k_wconv<<<gsz(LAT * KPMAX), 256, 0, stream>>>(P(widx), dst, K);
            }
    }

    // ---- one GMP block
    auto run_gmp = [&](int ge, const float* x, float* out, const float* r2,
                       const float* posb, const int* gi, const int* gj, int E, int N) {
        int base = exec_base[ge];
        auto WT = [&](int mlp, int ly) {
            return wt + (size_t)(ge * 6 + mlp * 3 + ly) * LAT * KPMAX;
        };
        int gE = (E + MT - 1) / MT, gN = (N + MT - 1) / MT;
        k_fill<<<gsz(N * LAT), 256, 0, stream>>>(aggr, 0.0f, N * LAT);
        // edge MLP (fused gather -> relu -> relu -> LN + segment_sum)
        k_mlp<<<gE, 256, 0, stream>>>(nullptr, x, posb, gi, gj, WT(0, 0), P(base + 1),
                                      nullptr, nullptr, nullptr, nullptr, nullptr,
                                      nullptr, act0, E, KPMAX, 1, 0);
        k_mlp<<<gE, 256, 0, stream>>>(act0, nullptr, nullptr, nullptr, nullptr,
                                      WT(0, 1), P(base + 3), nullptr, nullptr, nullptr,
                                      nullptr, nullptr, nullptr, act1, E, LAT, 0, 0);
        k_mlp<<<gE, 256, 0, stream>>>(act1, nullptr, nullptr, gi, nullptr,
                                      WT(0, 2), P(base + 5), P(base + 7), P(base + 6),
                                      aggr, nullptr, nullptr, nullptr, nullptr, E, LAT, 0, 1);
        // node MLP (fused concat -> relu -> relu -> LN + residual(s))
        k_mlp<<<gN, 256, 0, stream>>>(nullptr, x, aggr, nullptr, nullptr,
                                      WT(1, 0), P(base + 9), nullptr, nullptr, nullptr,
                                      nullptr, nullptr, nullptr, act0, N, 2 * LAT, 2, 0);
        k_mlp<<<gN, 256, 0, stream>>>(act0, nullptr, nullptr, nullptr, nullptr,
                                      WT(1, 1), P(base + 11), nullptr, nullptr, nullptr,
                                      nullptr, nullptr, nullptr, act1, N, LAT, 0, 0);
        k_mlp<<<gN, 256, 0, stream>>>(act1, nullptr, nullptr, nullptr, nullptr,
                                      WT(1, 2), P(base + 13), P(base + 15), P(base + 14),
                                      nullptr, x, r2, out, nullptr, N, LAT, 0, 2);
    };

    // ---- init level-0 state
    hipMemcpyAsync(hA, h_in, (size_t)N0 * LAT * 4, hipMemcpyDeviceToDevice, stream);
    hipMemcpyAsync(posA, pos_in, (size_t)N0 * 3 * 4, hipMemcpyDeviceToDevice, stream);
    k_fill<<<gsz(N0), 256, 0, stream>>>(wcur, 1.0f, N0);

    // ---- down path
    for (int l = 0; l < 2; ++l) {
        int N = Nlv[l], E = Ecnt[l], Nn = Nlv[l + 1];
        const int* gi = gs[l];
        const int* gj = gs[l] + E;
        hipMemcpyAsync(dposv[l], posA, (size_t)N * 3 * 4, hipMemcpyDeviceToDevice, stream);
        run_gmp(l, hA, hB, nullptr, posA, gi, gj, E, N);
        hipMemcpyAsync(doutv[l], hB, (size_t)N * LAT * 4, hipMemcpyDeviceToDevice, stream);
        // _cal_ew
        k_fill<<<gsz(N), 256, 0, stream>>>(deg, 0.0f, N);
        k_fill<<<gsz(N), 256, 0, stream>>>(waggr, 0.0f, N);
        k_deg<<<gsz(E), 256, 0, stream>>>(gi, deg, E);
        k_ws<<<gsz(E), 256, 0, stream>>>(gi, gj, wcur, deg, wse, waggr, E);
        k_ec<<<gsz(E), 256, 0, stream>>>(wse, waggr, gj, cts[l], E);
        // edge_conv (aggregating): h, pos
        k_fill<<<gsz(N * LAT), 256, 0, stream>>>(hU, 0.0f, N * LAT);
        k_econv<<<gsz(E * LAT), 256, 0, stream>>>(hB, gi, gj, cts[l], hU, E, LAT);
        k_fill<<<gsz(N * 3), 256, 0, stream>>>(posB, 0.0f, N * 3);
        k_econv<<<gsz(E * 3), 256, 0, stream>>>(posA, gi, gj, cts[l], posB, E, 3);
        // pool
        k_gather<<<gsz(Nn * LAT), 256, 0, stream>>>(hU, mids[l], hA, Nn, LAT);
        k_gather<<<gsz(Nn * 3), 256, 0, stream>>>(posB, mids[l], posA, Nn, 3);
        k_gather<<<gsz(Nn), 256, 0, stream>>>(waggr, mids[l], wcur, Nn, 1);
    }

    // ---- bottom
    run_gmp(2, hA, hB, nullptr, posA, gs[2], gs[2] + Ecnt[2], Ecnt[2], Nlv[2]);

    // ---- up path, d = 1
    {
        int N = Nlv[1], E = Ecnt[1];
        const int* gi = gs[1];
        const int* gj = gs[1] + E;
        k_fill<<<gsz(N * LAT), 256, 0, stream>>>(hU, 0.0f, N * LAT);
        k_scatter_rows<<<gsz(Nlv[2] * LAT), 256, 0, stream>>>(hB, mids[1], hU, Nlv[2], LAT);
        k_fill<<<gsz(N * LAT), 256, 0, stream>>>(hA, 0.0f, N * LAT);
        k_econv<<<gsz(E * LAT), 256, 0, stream>>>(hU, gj, gi, ct1, hA, E, LAT); // non-aggregating
        run_gmp(3, hA, hB, dout1, dpos1, gi, gj, E, N);
    }
    // ---- up path, d = 0  (writes final result to d_out)
    {
        int N = Nlv[0], E = Ecnt[0];
        const int* gi = gs[0];
        const int* gj = gs[0] + E;
        k_fill<<<gsz(N * LAT), 256, 0, stream>>>(hU, 0.0f, N * LAT);
        k_scatter_rows<<<gsz(Nlv[1] * LAT), 256, 0, stream>>>(hB, mids[0], hU, Nlv[1], LAT);
        k_fill<<<gsz(N * LAT), 256, 0, stream>>>(hA, 0.0f, N * LAT);
        k_econv<<<gsz(E * LAT), 256, 0, stream>>>(hU, gj, gi, ct0, hA, E, LAT);
        run_gmp(4, hA, (float*)d_out, dout0, dpos0, gi, gj, E, N);
    }
}
